// TensorProductWithMemoryOptimizationWithWeight_76046690943132
// MI455X (gfx1250) — matrix operand order repaired and checked by reading, not compiled, not run on hardware
//
#include <hip/hip_runtime.h>
#include <hip/hip_bf16.h>

// MI455X / gfx1250 fused tensor-product kernel.
// bf16 WMMA (v_wmma_f32_16x16x32_bf16) for all GEMMs, f32 accumulate.
// One wave processes 16 rows end-to-end; 8 waves / workgroup.
// Weights converted once per workgroup to bf16 LDS, transposed (n-major,
// K-contiguous) so B fragments are contiguous 16-byte LDS loads.
// A fragments are hoisted out of column-tile loops (loaded once per GEMM).

typedef __attribute__((ext_vector_type(16))) __bf16 bf16x16;
typedef __attribute__((ext_vector_type(8)))  __bf16 bf16x8;
typedef __attribute__((ext_vector_type(8)))  float  f32x8;

// ---- LDS layout (bytes) ----
#define WOFF_TP1   0        // 64x64  bf16 transposed (ldk=64)
#define WOFF_TP2   8192
#define WOFF_TP3   16384
#define WOFF_TP4   24576
#define WOFF_FC1   32768
#define WOFF_FC2   40960
#define WOFF_FC3   49152    // 256x64 bf16 transposed (ldk=64)
#define WOFF_LIN0  81920    // 64x128 bf16 transposed (ldk=128)
#define WOFF_LIN1  98304
#define WAVE_BASE  114688
#define WAVE_BYTES 24832
// per-wave: st0(2048) st1(2048) sst(4096) t2b(2048) wb(8192) a3st(6144) yv(256)
#define SMEM_TOTAL (WAVE_BASE + 8 * WAVE_BYTES)   // 313344 B <= 320KB/WGP

// A fragment: 16x32 bf16, row-major source with row stride ldk (elements).
// Layout used here (contiguous k): lanes 0-15 row M=lane, elems 0..15 -> K=kb+0..15
//             lanes 16-31: same rows, K=kb+16..31.
__device__ __forceinline__ bf16x16 load_a_frag(const __bf16* src, int ldk,
                                               int kbase, int lane) {
  const int half = lane >> 4;
  const int row  = lane & 15;
  const __bf16* p = src + row * ldk + kbase + half * 16;
  bf16x8 lo = *(const bf16x8*)(p);
  bf16x8 hi = *(const bf16x8*)(p + 8);
  return __builtin_shufflevector(lo, hi, 0, 1, 2, 3, 4, 5, 6, 7,
                                 8, 9, 10, 11, 12, 13, 14, 15);
}

// B fragment: 32x16 bf16 from transposed weight (wt[n][k], row stride ldk).
// Layout used here (contiguous k, as A): lane -> column N=lane%16; lanes 0-15 hold K=kb+0..15,
//             lanes 16-31 hold K=kb+16..31 (K ascending through elements).
__device__ __forceinline__ bf16x16 load_b_frag(const __bf16* wt, int ldk,
                                               int kbase, int nbase, int lane) {
  const int col = lane & 15;
  const int kh  = (lane >> 4) * 16;
  return *(const bf16x16*)(wt + (nbase + col) * ldk + kbase + kh);
}

__device__ __forceinline__ f32x8 wmma2(bf16x16 a0, bf16x16 a1,
                                       bf16x16 b0, bf16x16 b1) {
  f32x8 c = {0.f, 0.f, 0.f, 0.f, 0.f, 0.f, 0.f, 0.f};
  c = __builtin_amdgcn_wmma_f32_16x16x32_bf16(false, a0, false, b0,
                                              (short)0, c, false, false);
  c = __builtin_amdgcn_wmma_f32_16x16x32_bf16(false, a1, false, b1,
                                              (short)0, c, false, false);
  return c;
}

__device__ __forceinline__ float silu(float v) {
  // v * sigmoid(v) via hardware v_rcp_f32 (avoid IEEE div macro expansion)
  return v * __builtin_amdgcn_rcpf(1.f + __expf(-v));
}

template <int K, int Nw>
__device__ __forceinline__ void conv_w(const float* __restrict__ src,
                                       __bf16* dst, int tid) {
#pragma unroll 1
  for (int idx = tid; idx < K * Nw; idx += 256) {
    int k = idx / Nw, c = idx & (Nw - 1);
    dst[c * K + k] = (__bf16)src[idx];
  }
}

__global__ void __launch_bounds__(256)
fused_tp_kernel(const float* __restrict__ x, const float* __restrict__ y,
                const float* __restrict__ sc,
                const float* __restrict__ Wtp1, const float* __restrict__ Wtp2,
                const float* __restrict__ Wtp3, const float* __restrict__ Wtp4,
                const float* __restrict__ Wfc1, const float* __restrict__ Wfc2,
                const float* __restrict__ Wfc3, const float* __restrict__ Wlin0,
                const float* __restrict__ Wlin1,
                float* __restrict__ out, int n) {
  extern __shared__ char smem[];
  const int tid = threadIdx.x;

  // ---- convert all weights to bf16, transposed: Wt[col][k] = W[k][col] ----
  conv_w<64, 64>(Wtp1,  (__bf16*)(smem + WOFF_TP1),  tid);
  conv_w<64, 64>(Wtp2,  (__bf16*)(smem + WOFF_TP2),  tid);
  conv_w<64, 64>(Wtp3,  (__bf16*)(smem + WOFF_TP3),  tid);
  conv_w<64, 64>(Wtp4,  (__bf16*)(smem + WOFF_TP4),  tid);
  conv_w<64, 64>(Wfc1,  (__bf16*)(smem + WOFF_FC1),  tid);
  conv_w<64, 64>(Wfc2,  (__bf16*)(smem + WOFF_FC2),  tid);
  conv_w<64, 256>(Wfc3, (__bf16*)(smem + WOFF_FC3),  tid);
  conv_w<128, 64>(Wlin0,(__bf16*)(smem + WOFF_LIN0), tid);
  conv_w<128, 64>(Wlin1,(__bf16*)(smem + WOFF_LIN1), tid);
  __syncthreads();

  const int lane = tid & 31;
  const int wave = tid >> 5;
  char* wbase = smem + WAVE_BASE + wave * WAVE_BYTES;
  __bf16* st0  = (__bf16*)(wbase);            // 16x64 A staging (ping)
  __bf16* st1  = (__bf16*)(wbase + 2048);     // 16x64 A staging (pong)
  __bf16* sst  = (__bf16*)(wbase + 4096);     // 16x128 gated s0/s1 staging
  __bf16* t2b  = (__bf16*)(wbase + 8192);     // 16x64  t2 = x0@W_tp2
  __bf16* wb   = (__bf16*)(wbase + 10240);    // 16x256 MLP gate w
  __bf16* a3st = (__bf16*)(wbase + 18432);    // 16x192 a3_m = x1[:,:,m]*y0
  float*  yv   = (float*)(wbase + 24576);     // 16x4

  const __bf16* wt_tp1  = (const __bf16*)(smem + WOFF_TP1);
  const __bf16* wt_tp2  = (const __bf16*)(smem + WOFF_TP2);
  const __bf16* wt_tp3  = (const __bf16*)(smem + WOFF_TP3);
  const __bf16* wt_tp4  = (const __bf16*)(smem + WOFF_TP4);
  const __bf16* wt_fc1  = (const __bf16*)(smem + WOFF_FC1);
  const __bf16* wt_fc2  = (const __bf16*)(smem + WOFF_FC2);
  const __bf16* wt_fc3  = (const __bf16*)(smem + WOFF_FC3);
  const __bf16* wt_lin0 = (const __bf16*)(smem + WOFF_LIN0);
  const __bf16* wt_lin1 = (const __bf16*)(smem + WOFF_LIN1);

  const int row0  = (blockIdx.x * 8 + wave) * 16;
  const int mhalf = (lane >> 4) * 8;
  const int ncl   = lane & 15;

  const float inv8 = 0.125f;                   // 1/sqrt(64)
  const float cmul = 0.125f;                   // 1/sqrt(MUL)
  const float cmid = 0.08838834764831845f;     // 1/sqrt(128)
  const float is3  = 0.5773502691896258f;      // 1/sqrt(3)

  // ---- stage y and scalars ----
  for (int i = lane; i < 64; i += 32) {
    int r = i >> 2, c = i & 3, row = row0 + r;
    yv[i] = (row < n) ? y[row * 4 + c] : 0.f;
  }
  for (int i = lane; i < 16 * 64; i += 32) {
    int r = i >> 6, row = row0 + r;
    st0[i] = (row < n) ? (__bf16)sc[row * 64 + (i & 63)] : (__bf16)0.f;
  }
  __syncthreads();

  // ---- MLP: h1 = silu(sc@Wfc1/8) ----
  {
    bf16x16 a0 = load_a_frag(st0, 64, 0, lane);
    bf16x16 a1 = load_a_frag(st0, 64, 32, lane);
#pragma unroll
    for (int ct = 0; ct < 4; ++ct) {
      bf16x16 b0 = load_b_frag(wt_fc1, 64, 0, ct * 16, lane);
      bf16x16 b1 = load_b_frag(wt_fc1, 64, 32, ct * 16, lane);
      f32x8 c = wmma2(a0, a1, b0, b1);
      int Ncol = ct * 16 + ncl;
#pragma unroll
      for (int rI = 0; rI < 8; ++rI)
        st1[(rI + mhalf) * 64 + Ncol] = (__bf16)silu(c[rI] * inv8);
    }
  }
  __syncthreads();

  // ---- h2 = silu(h1@Wfc2/8) ----
  {
    bf16x16 a0 = load_a_frag(st1, 64, 0, lane);
    bf16x16 a1 = load_a_frag(st1, 64, 32, lane);
#pragma unroll
    for (int ct = 0; ct < 4; ++ct) {
      bf16x16 b0 = load_b_frag(wt_fc2, 64, 0, ct * 16, lane);
      bf16x16 b1 = load_b_frag(wt_fc2, 64, 32, ct * 16, lane);
      f32x8 c = wmma2(a0, a1, b0, b1);
      int Ncol = ct * 16 + ncl;
#pragma unroll
      for (int rI = 0; rI < 8; ++rI)
        st0[(rI + mhalf) * 64 + Ncol] = (__bf16)silu(c[rI] * inv8);
    }
  }
  __syncthreads();

  // ---- w = h2@Wfc3/8   (N=256) ----
  {
    bf16x16 a0 = load_a_frag(st0, 64, 0, lane);
    bf16x16 a1 = load_a_frag(st0, 64, 32, lane);
#pragma unroll 4
    for (int ct = 0; ct < 16; ++ct) {
      bf16x16 b0 = load_b_frag(wt_fc3, 64, 0, ct * 16, lane);
      bf16x16 b1 = load_b_frag(wt_fc3, 64, 32, ct * 16, lane);
      f32x8 c = wmma2(a0, a1, b0, b1);
      int Ncol = ct * 16 + ncl;
#pragma unroll
      for (int rI = 0; rI < 8; ++rI)
        wb[(rI + mhalf) * 256 + Ncol] = (__bf16)(c[rI] * inv8);
    }
  }
  __syncthreads();

  // ---- stage a0 = x0*y0, a4 = (x1.y1)/sqrt3, a3_m = x1_m*y0 ----
  for (int i = lane; i < 16 * 64; i += 32) {
    int r = i >> 6, u = i & 63, row = row0 + r;
    float y0 = yv[r * 4 + 0];
    float x0v = 0.f, a4v = 0.f, a30 = 0.f, a31 = 0.f, a32 = 0.f;
    if (row < n) {
      const float* xp = x + (long long)row * 256;
      x0v = xp[u];
      float x10 = xp[64 + 3 * u], x11 = xp[64 + 3 * u + 1], x12 = xp[64 + 3 * u + 2];
      a4v = (x10 * yv[r * 4 + 1] + x11 * yv[r * 4 + 2] + x12 * yv[r * 4 + 3]) * is3;
      a30 = x10 * y0; a31 = x11 * y0; a32 = x12 * y0;
    }
    st0[i] = (__bf16)(x0v * y0);
    st1[i] = (__bf16)a4v;
    a3st[r * 192 + u]       = (__bf16)a30;
    a3st[r * 192 + 64 + u]  = (__bf16)a31;
    a3st[r * 192 + 128 + u] = (__bf16)a32;
  }
  __syncthreads();

  // ---- s0 = [a0@Wtp1, a4@Wtp4]*cmul * w0 -> sst (bf16) ----
  {
    bf16x16 aa0 = load_a_frag(st0, 64, 0, lane);
    bf16x16 aa1 = load_a_frag(st0, 64, 32, lane);
    bf16x16 ab0 = load_a_frag(st1, 64, 0, lane);
    bf16x16 ab1 = load_a_frag(st1, 64, 32, lane);
#pragma unroll 2
    for (int ct = 0; ct < 4; ++ct) {
      bf16x16 b0 = load_b_frag(wt_tp1, 64, 0, ct * 16, lane);
      bf16x16 b1 = load_b_frag(wt_tp1, 64, 32, ct * 16, lane);
      f32x8 ca = wmma2(aa0, aa1, b0, b1);
      bf16x16 b2 = load_b_frag(wt_tp4, 64, 0, ct * 16, lane);
      bf16x16 b3 = load_b_frag(wt_tp4, 64, 32, ct * 16, lane);
      f32x8 cb = wmma2(ab0, ab1, b2, b3);
      int Ncol = ct * 16 + ncl;
#pragma unroll
      for (int rI = 0; rI < 8; ++rI) {
        int M = rI + mhalf;
        float w0a = (float)wb[M * 256 + Ncol];
        float w0b = (float)wb[M * 256 + 64 + Ncol];
        sst[M * 128 + Ncol]      = (__bf16)(ca[rI] * cmul * w0a);
        sst[M * 128 + 64 + Ncol] = (__bf16)(cb[rI] * cmul * w0b);
      }
    }
  }
  __syncthreads();

  // ---- out0 = s0 @ Wlin0 * cmid ----
  {
    bf16x16 a0 = load_a_frag(sst, 128, 0, lane);
    bf16x16 a1 = load_a_frag(sst, 128, 32, lane);
    bf16x16 a2 = load_a_frag(sst, 128, 64, lane);
    bf16x16 a3 = load_a_frag(sst, 128, 96, lane);
#pragma unroll 2
    for (int ct = 0; ct < 4; ++ct) {
      bf16x16 b0 = load_b_frag(wt_lin0, 128, 0, ct * 16, lane);
      bf16x16 b1 = load_b_frag(wt_lin0, 128, 32, ct * 16, lane);
      bf16x16 b2 = load_b_frag(wt_lin0, 128, 64, ct * 16, lane);
      bf16x16 b3 = load_b_frag(wt_lin0, 128, 96, ct * 16, lane);
      f32x8 c = wmma2(a0, a1, b0, b1);
      c = __builtin_amdgcn_wmma_f32_16x16x32_bf16(false, a2, false, b2,
                                                  (short)0, c, false, false);
      c = __builtin_amdgcn_wmma_f32_16x16x32_bf16(false, a3, false, b3,
                                                  (short)0, c, false, false);
      int Ncol = ct * 16 + ncl;
#pragma unroll
      for (int rI = 0; rI < 8; ++rI) {
        int row = row0 + rI + mhalf;
        if (row < n) out[(long long)row * 256 + Ncol] = c[rI] * cmid;
      }
    }
  }

  // ---- stage a2 = x0 ----
  for (int i = lane; i < 16 * 64; i += 32) {
    int r = i >> 6, row = row0 + r;
    st0[i] = (row < n) ? (__bf16)x[(long long)row * 256 + (i & 63)] : (__bf16)0.f;
  }
  __syncthreads();

  // ---- t2 = x0 @ Wtp2 (unscaled, bf16) ----
  {
    bf16x16 a0 = load_a_frag(st0, 64, 0, lane);
    bf16x16 a1 = load_a_frag(st0, 64, 32, lane);
#pragma unroll
    for (int ct = 0; ct < 4; ++ct) {
      bf16x16 b0 = load_b_frag(wt_tp2, 64, 0, ct * 16, lane);
      bf16x16 b1 = load_b_frag(wt_tp2, 64, 32, ct * 16, lane);
      f32x8 c = wmma2(a0, a1, b0, b1);
      int Ncol = ct * 16 + ncl;
#pragma unroll
      for (int rI = 0; rI < 8; ++rI)
        t2b[(rI + mhalf) * 64 + Ncol] = (__bf16)c[rI];
    }
  }
  __syncthreads();

  // ---- per m: s1_m -> out1_m ----
#pragma unroll 1
  for (int m = 0; m < 3; ++m) {
    // mid1b half: (a3_m @ Wtp3)*cmul * w1[64:]
    {
      bf16x16 a0 = load_a_frag(a3st, 192, m * 64, lane);
      bf16x16 a1 = load_a_frag(a3st, 192, m * 64 + 32, lane);
#pragma unroll
      for (int ct = 0; ct < 4; ++ct) {
        bf16x16 b0 = load_b_frag(wt_tp3, 64, 0, ct * 16, lane);
        bf16x16 b1 = load_b_frag(wt_tp3, 64, 32, ct * 16, lane);
        f32x8 c = wmma2(a0, a1, b0, b1);
        int Ncol = ct * 16 + ncl;
#pragma unroll
        for (int rI = 0; rI < 8; ++rI) {
          int M = rI + mhalf;
          float w1v = (float)wb[M * 256 + 128 + 64 + Ncol];
          sst[M * 128 + 64 + Ncol] = (__bf16)(c[rI] * cmul * w1v);
        }
      }
    }
    // mid1a half: t2 * y1[m] * cmul * w1[:64]
    for (int i = lane; i < 16 * 64; i += 32) {
      int r = i >> 6, k2 = i & 63;
      float w1v = (float)wb[r * 256 + 128 + k2];
      sst[r * 128 + k2] =
          (__bf16)((float)t2b[r * 64 + k2] * yv[r * 4 + 1 + m] * cmul * w1v);
    }
    __syncthreads();
    // out1_m = s1_m @ Wlin1 * cmid ; out col = 64 + 3*u + m
    {
      bf16x16 a0 = load_a_frag(sst, 128, 0, lane);
      bf16x16 a1 = load_a_frag(sst, 128, 32, lane);
      bf16x16 a2 = load_a_frag(sst, 128, 64, lane);
      bf16x16 a3 = load_a_frag(sst, 128, 96, lane);
#pragma unroll 2
      for (int ct = 0; ct < 4; ++ct) {
        bf16x16 b0 = load_b_frag(wt_lin1, 128, 0, ct * 16, lane);
        bf16x16 b1 = load_b_frag(wt_lin1, 128, 32, ct * 16, lane);
        bf16x16 b2 = load_b_frag(wt_lin1, 128, 64, ct * 16, lane);
        bf16x16 b3 = load_b_frag(wt_lin1, 128, 96, ct * 16, lane);
        f32x8 c = wmma2(a0, a1, b0, b1);
        c = __builtin_amdgcn_wmma_f32_16x16x32_bf16(false, a2, false, b2,
                                                    (short)0, c, false, false);
        c = __builtin_amdgcn_wmma_f32_16x16x32_bf16(false, a3, false, b3,
                                                    (short)0, c, false, false);
        int u = ct * 16 + ncl;
#pragma unroll
        for (int rI = 0; rI < 8; ++rI) {
          int row = row0 + rI + mhalf;
          if (row < n) out[(long long)row * 256 + 64 + 3 * u + m] = c[rI] * cmid;
        }
      }
    }
    __syncthreads();
  }
}

extern "C" void kernel_launch(void* const* d_in, const int* in_sizes, int n_in,
                              void* d_out, int out_size, void* d_ws, size_t ws_size,
                              hipStream_t stream) {
  const float* x     = (const float*)d_in[0];
  const float* y     = (const float*)d_in[1];
  const float* sc    = (const float*)d_in[2];
  const float* Wtp1  = (const float*)d_in[3];
  const float* Wtp2  = (const float*)d_in[4];
  const float* Wtp3  = (const float*)d_in[5];
  const float* Wtp4  = (const float*)d_in[6];
  const float* Wfc1  = (const float*)d_in[7];
  const float* Wfc2  = (const float*)d_in[8];
  const float* Wfc3  = (const float*)d_in[9];
  const float* Wlin0 = (const float*)d_in[10];
  const float* Wlin1 = (const float*)d_in[11];
  float* out = (float*)d_out;

  const int n = in_sizes[0] / 256;           // x is (N, 256)
  const int rows_per_wg = 8 * 16;            // 8 waves x 16 rows
  const int grid = (n + rows_per_wg - 1) / rows_per_wg;
  fused_tp_kernel<<<grid, 256, SMEM_TOTAL, stream>>>(
      x, y, sc, Wtp1, Wtp2, Wtp3, Wtp4, Wfc1, Wfc2, Wfc3, Wlin0, Wlin1, out, n);
}
